// GINConv_8856222564747
// MI455X (gfx1250) — matrix operand order repaired and checked by reading, not compiled, not run on hardware
//
#include <hip/hip_runtime.h>

#define N_NODES 10000
#define D_IN    512
#define D_HID   1024
#define N_EDGES 160000

typedef __attribute__((ext_vector_type(16))) __bf16 v16bf;
typedef __attribute__((ext_vector_type(8)))  __bf16 v8bf;
typedef __attribute__((ext_vector_type(8)))  float  v8f;

// ---------------- Stage 1: rst = (1+eps)*feat ----------------
__global__ void k_init_rst(const float* __restrict__ feat,
                           const float* __restrict__ eps,
                           float* __restrict__ rst, int n4) {
    int i = blockIdx.x * blockDim.x + threadIdx.x;
    if (i < n4) {
        float s = 1.0f + eps[0];
        float4 v = ((const float4*)feat)[i];
        float4 o;
        o.x = s * v.x; o.y = s * v.y; o.z = s * v.z; o.w = s * v.w;
        ((float4*)rst)[i] = o;
    }
}

// ---------------- Stage 2: scatter-add over edges ----------------
// one thread per (edge, float4-chunk); D_IN/4 = 128 chunks per edge
__global__ void k_scatter(const float* __restrict__ feat,
                          const int* __restrict__ src,
                          const int* __restrict__ dst,
                          float* __restrict__ rst) {
    int t = blockIdx.x * blockDim.x + threadIdx.x;
    int e = t >> 7;          // / 128
    int c = t & 127;
    if (e < N_EDGES) {
        int s = src[e];
        int d = dst[e];
        float4 v = ((const float4*)(feat + (size_t)s * D_IN))[c];
        float* p = rst + (size_t)d * D_IN + (c << 2);
        atomicAdd(p + 0, v.x);
        atomicAdd(p + 1, v.y);
        atomicAdd(p + 2, v.z);
        atomicAdd(p + 3, v.w);
    }
}

// ---------------- f32 -> bf16 convert ----------------
__global__ void k_cvt(const float* __restrict__ in, __bf16* __restrict__ out, int n) {
    int i = blockIdx.x * blockDim.x + threadIdx.x;
    if (i < n) out[i] = (__bf16)in[i];
}

// ---------------- f32 [K,N] -> bf16 transposed [N,K] ----------------
__global__ void k_cvt_t(const float* __restrict__ W, __bf16* __restrict__ Wt,
                        int K, int N) {
    int i = blockIdx.x * blockDim.x + threadIdx.x;
    if (i < K * N) {
        int k = i / N;
        int n = i - k * N;
        Wt[(size_t)n * K + k] = (__bf16)W[i];
    }
}

// A fragment: lane half 0 -> elems {K0..7, K16..23}; half 1 -> {K8..15, K24..31}
__device__ __forceinline__ v16bf load_fragA(const __bf16* p, int kbA) {
    v8bf x0 = *(const v8bf*)(p + kbA);
    v8bf x1 = *(const v8bf*)(p + 16 + kbA);
    return __builtin_shufflevector(x0, x1,
              0,1,2,3,4,5,6,7,8,9,10,11,12,13,14,15);
}

// B fragment: lane half 0 -> elems {K0..7, K16..23}; half 1 -> {K8..15, K24..31}
__device__ __forceinline__ v16bf load_fragB(const __bf16* p, int kbB) {
    v8bf x0 = *(const v8bf*)(p + kbB);
    v8bf x1 = *(const v8bf*)(p + 16 + kbB);
    return __builtin_shufflevector(x0, x1,
              0,1,2,3,4,5,6,7,8,9,10,11,12,13,14,15);
}

// ---- WMMA GEMM, 2x2 register-blocked: one wave -> 32x32 output tile ----
// out = act(A[M,K] @ Bt[N,K]^T + bias) (+resid). K mult of 32, N mult of 32.
// M tail (10000 % 32 == 16) handled by guarded stores; OOB A reads stay in ws.
__global__ void k_gemm(const __bf16* __restrict__ A,
                       const __bf16* __restrict__ Bt,
                       const float*  __restrict__ bias,
                       const float*  __restrict__ resid,
                       float*  __restrict__ outF,
                       __bf16* __restrict__ outB,
                       int M, int N, int K, int relu) {
    int wid = blockIdx.x * (blockDim.x >> 5) + (threadIdx.x >> 5);
    int ntN = N >> 5;                          // 32-wide col tiles
    int nTM = (M + 31) >> 5;                   // 32-tall row tiles (ceil)
    int tM  = wid / ntN;
    int tN  = wid - tM * ntN;
    if (tM >= nTM) return;                     // wave-uniform

    int lane = threadIdx.x & 31;
    int half = lane >> 4;
    int l16  = lane & 15;
    int kbA  = half * 8;
    int kbB  = half * 8;

    const __bf16* pa0 = A  + (size_t)(tM * 32 + l16) * K;
    const __bf16* pa1 = pa0 + (size_t)16 * K;
    const __bf16* pb0 = Bt + (size_t)(tN * 32 + l16) * K;
    const __bf16* pb1 = pb0 + (size_t)16 * K;

    v8f acc00 = {0,0,0,0,0,0,0,0};
    v8f acc01 = {0,0,0,0,0,0,0,0};
    v8f acc10 = {0,0,0,0,0,0,0,0};
    v8f acc11 = {0,0,0,0,0,0,0,0};

    for (int k0 = 0; k0 < K; k0 += 32) {
        v16bf a0 = load_fragA(pa0 + k0, kbA);
        v16bf a1 = load_fragA(pa1 + k0, kbA);
        v16bf b0 = load_fragB(pb0 + k0, kbB);
        v16bf b1 = load_fragB(pb1 + k0, kbB);
        acc00 = __builtin_amdgcn_wmma_f32_16x16x32_bf16(
                    false, a0, false, b0, (short)0, acc00, false, false);
        acc01 = __builtin_amdgcn_wmma_f32_16x16x32_bf16(
                    false, a0, false, b1, (short)0, acc01, false, false);
        acc10 = __builtin_amdgcn_wmma_f32_16x16x32_bf16(
                    false, a1, false, b0, (short)0, acc10, false, false);
        acc11 = __builtin_amdgcn_wmma_f32_16x16x32_bf16(
                    false, a1, false, b1, (short)0, acc11, false, false);
    }

    // D layout per 16x16 tile: VGPR r holds M = mbase + half*8 + r, N = col
    int col0 = tN * 32 + l16;
    int col1 = col0 + 16;
    float bn0 = bias ? bias[col0] : 0.0f;
    float bn1 = bias ? bias[col1] : 0.0f;

#pragma unroll
    for (int i = 0; i < 2; ++i) {
        int mbase = tM * 32 + i * 16 + half * 8;
        v8f ac0 = i ? acc10 : acc00;
        v8f ac1 = i ? acc11 : acc01;
#pragma unroll
        for (int r = 0; r < 8; ++r) {
            int m = mbase + r;
            if (m < M) {
                float v0 = ac0[r] + bn0;
                float v1 = ac1[r] + bn1;
                if (relu) { v0 = fmaxf(v0, 0.0f); v1 = fmaxf(v1, 0.0f); }
                size_t i0 = (size_t)m * N + col0;
                size_t i1 = (size_t)m * N + col1;
                if (resid) { v0 += resid[i0]; v1 += resid[i1]; }
                if (outF) { outF[i0] = v0; outF[i1] = v1; }
                if (outB) { outB[i0] = (__bf16)v0; outB[i1] = (__bf16)v1; }
            }
        }
    }
}

extern "C" void kernel_launch(void* const* d_in, const int* in_sizes, int n_in,
                              void* d_out, int out_size, void* d_ws, size_t ws_size,
                              hipStream_t stream) {
    const float* feat = (const float*)d_in[0];
    const float* W1   = (const float*)d_in[1];
    const float* b1   = (const float*)d_in[2];
    const float* W2   = (const float*)d_in[3];
    const float* b2   = (const float*)d_in[4];
    const float* eps  = (const float*)d_in[5];
    const int*   src  = (const int*)d_in[6];
    const int*   dst  = (const int*)d_in[7];
    float* out = (float*)d_out;

    // workspace layout (bytes):
    //   [0, 20.48M)      rst f32  (later reused as h bf16 [N_NODES, D_HID])
    //   [20.48M, 30.72M) rst bf16
    //   [30.72M, +1M)    W1^T bf16  [D_HID, D_IN]
    //   [31.77M, +1M)    W2^T bf16  [D_IN, D_HID]
    char* ws = (char*)d_ws;
    float*  rstF = (float*)(ws + 0);
    __bf16* hb   = (__bf16*)(ws + 0);                  // aliases rstF (dead by then)
    __bf16* rstb = (__bf16*)(ws + 20480000);
    __bf16* W1t  = (__bf16*)(ws + 30720000);
    __bf16* W2t  = (__bf16*)(ws + 31768576);

    // 1) rst = (1+eps)*feat
    int n4 = N_NODES * D_IN / 4;
    k_init_rst<<<(n4 + 255) / 256, 256, 0, stream>>>(feat, eps, rstF, n4);

    // 2) scatter-add neighbors (f32 atomics, L2-resident accumulator)
    int nt = N_EDGES * (D_IN / 4);
    k_scatter<<<nt / 256, 256, 0, stream>>>(feat, src, dst, rstF);

    // 3) weight converts (transposed, K-contiguous for WMMA B loads)
    k_cvt_t<<<(D_IN * D_HID + 255) / 256, 256, 0, stream>>>(W1, W1t, D_IN, D_HID);
    k_cvt_t<<<(D_HID * D_IN + 255) / 256, 256, 0, stream>>>(W2, W2t, D_HID, D_IN);

    // 4) rst -> bf16
    k_cvt<<<(N_NODES * D_IN + 255) / 256, 256, 0, stream>>>(rstF, rstb, N_NODES * D_IN);

    // 5) h = relu(rst @ W1 + b1)   [10000 x 1024] bf16
    int nTM = (N_NODES + 31) / 32;                     // 313
    int waves1 = nTM * (D_HID / 32);                   // 313*32 = 10016
    k_gemm<<<(waves1 + 7) / 8, 256, 0, stream>>>(rstb, W1t, b1, nullptr,
                                                 nullptr, hb,
                                                 N_NODES, D_HID, D_IN, 1);

    // 6) out = h @ W2 + b2 + feat  [10000 x 512] f32
    int waves2 = nTM * (D_IN / 32);                    // 313*16 = 5008
    k_gemm<<<(waves2 + 7) / 8, 256, 0, stream>>>(hb, W2t, b2, feat,
                                                 out, nullptr,
                                                 N_NODES, D_IN, D_HID, 0);
}
